// FlashAttentionScratch_69879117906643
// MI455X (gfx1250) — matrix operand order repaired and checked by reading, not compiled, not run on hardware
//
#include <hip/hip_runtime.h>

// ---------------------------------------------------------------- constants
#define Bq    2
#define Sq    2048
#define Dm    1024
#define Hh    16
#define HDim  64
#define Mrows (Bq * Sq)   // 4096

typedef __bf16 bf16_t;
typedef __attribute__((ext_vector_type(16))) __bf16 v16bf;
typedef __attribute__((ext_vector_type(8)))  float  v8f;

union FragBF { v16bf v; uint4 q[2]; };

__device__ __forceinline__ bf16_t f2bf(float f) {
  // round-to-nearest-even fp32 -> bf16
  unsigned u = __builtin_bit_cast(unsigned, f);
  unsigned r = u + 0x7FFFu + ((u >> 16) & 1u);
  unsigned short s = (unsigned short)(r >> 16);
  return __builtin_bit_cast(bf16_t, s);
}

// A-fragment (16x32, M x K): lanes 0-15 hold K = k..k+15,
// lanes 16-31 hold K = k+16..k+31 (same k order as B-fragment).
__device__ __forceinline__ v16bf load_frag_a(const bf16_t* p, int khalf) {
  FragBF f;
  f.q[0] = *(const uint4*)(p + 2 * khalf);
  f.q[1] = *(const uint4*)(p + 2 * khalf + 8);
  return f.v;
}

// B-fragment (32x16, K x N): lane n = column; lanes 0-15 hold K = k..k+15,
// lanes 16-31 hold K = k+16..k+31, 2 values per VGPR (contiguous k order, same as A-fragment).
__device__ __forceinline__ v16bf load_frag_b(const bf16_t* p, int k16) {
  FragBF f;
  f.q[0] = *(const uint4*)(p + k16);
  f.q[1] = *(const uint4*)(p + k16 + 8);
  return f.v;
}

__device__ __forceinline__ v8f wmma_bf16(v16bf a, v16bf b, v8f c) {
  return __builtin_amdgcn_wmma_f32_16x16x32_bf16(
      /*neg_a=*/false, a, /*neg_b=*/false, b,
      /*c_mod=*/(short)0, c, /*reuse_a=*/false, /*reuse_b=*/false);
}

// CDNA5 async copy: 16 bytes global -> LDS, tracked by ASYNCcnt (ISA 10.x).
// lds_off is the low-32-bit LDS byte address (generic ptr truncates to LDS
// offset per ISA 10.2 aperture rules).
__device__ __forceinline__ void async_ld16(unsigned lds_off, const bf16_t* g) {
  asm volatile("global_load_async_to_lds_b128 %0, %1, off"
               :: "v"(lds_off), "v"(g)
               : "memory");
}
__device__ __forceinline__ void wait_async0() {
  asm volatile("s_wait_asynccnt 0x0" ::: "memory");
}

// ---------------------------------------------------------------- fp32 -> bf16
__global__ void k_cvt(const float* __restrict__ in, bf16_t* __restrict__ out, int n) {
  int i = (blockIdx.x * blockDim.x + threadIdx.x) * 4;
  if (i >= n) return;
  float4 f = *(const float4*)(in + i);
  union { bf16_t h[4]; uint2 u; } p;
  p.h[0] = f2bf(f.x); p.h[1] = f2bf(f.y); p.h[2] = f2bf(f.z); p.h[3] = f2bf(f.w);
  *(uint2*)(out + i) = p.u;
}

// ------------------------------------------------- W[k,n] fp32 -> Wt[n,k] bf16
__global__ void k_transpose_cvt(const float* __restrict__ W, bf16_t* __restrict__ Wt) {
  __shared__ float tile[32][33];
  int bx = blockIdx.x * 32, by = blockIdx.y * 32;
  int tx = threadIdx.x, ty = threadIdx.y;            // 32 x 8
  for (int r = 0; r < 32; r += 8)
    tile[ty + r][tx] = W[(size_t)(by + ty + r) * Dm + bx + tx];
  __syncthreads();
  for (int r = 0; r < 32; r += 8)
    Wt[(size_t)(bx + ty + r) * Dm + by + tx] = f2bf(tile[tx][ty + r]);
}

// ------------------------------------------------------------- bf16 GEMM
// C[M,N] = A[M,K] * Bt[N,K]^T.  mode 0: out bf16 [B,H,S,HD] (Q/K layout)
// mode 1: out bf16 [B,H,HD,S] (V transposed)  mode 2: fp32 out + bias
__global__ void k_gemm_bf16(const bf16_t* __restrict__ A, const bf16_t* __restrict__ Bt,
                            bf16_t* __restrict__ outb, float* __restrict__ outf,
                            const float* __restrict__ bias, int mode) {
  const int tid  = threadIdx.x;
  const int lane = tid & 31;
  const int w    = tid >> 5;           // 8 waves
  const int rlo  = lane & 15;
  const int hi   = lane >> 4;
  const int khalfA = hi * 8;
  const int k16B   = hi * 16;

  const int mBase = blockIdx.y * 128 + (w & 3) * 32;   // wave grid 4 x 2
  const int nBase = blockIdx.x * 64  + (w >> 2) * 32;

  const v8f vzero = {0.f,0.f,0.f,0.f,0.f,0.f,0.f,0.f};
  v8f c[2][2];
  c[0][0] = vzero; c[0][1] = vzero; c[1][0] = vzero; c[1][1] = vzero;

  const bf16_t* a0p = A  + (size_t)(mBase      + rlo) * Dm;
  const bf16_t* a1p = A  + (size_t)(mBase + 16 + rlo) * Dm;
  const bf16_t* b0p = Bt + (size_t)(nBase      + rlo) * Dm;
  const bf16_t* b1p = Bt + (size_t)(nBase + 16 + rlo) * Dm;

  for (int k = 0; k < Dm; k += 32) {
    v16bf a0 = load_frag_a(a0p + k, khalfA);
    v16bf a1 = load_frag_a(a1p + k, khalfA);
    v16bf b0 = load_frag_b(b0p + k, k16B);
    v16bf b1 = load_frag_b(b1p + k, k16B);
    c[0][0] = wmma_bf16(a0, b0, c[0][0]);
    c[0][1] = wmma_bf16(a0, b1, c[0][1]);
    c[1][0] = wmma_bf16(a1, b0, c[1][0]);
    c[1][1] = wmma_bf16(a1, b1, c[1][1]);
  }

  // C/D layout: VGPR r, lanes 0-15 -> M=r, lanes 16-31 -> M=8+r; N = rlo.
  for (int i = 0; i < 2; ++i)
    for (int j = 0; j < 2; ++j)
      for (int r = 0; r < 8; ++r) {
        int row = mBase + i * 16 + r + hi * 8;
        int col = nBase + j * 16 + rlo;
        float v = c[i][j][r];
        if (mode == 2) {
          outf[(size_t)row * Dm + col] = v + bias[col];
        } else {
          int b = row >> 11, s = row & (Sq - 1);
          int h = col >> 6,  hd = col & (HDim - 1);
          size_t idx = (mode == 0)
              ? ((((size_t)b * Hh + h) * Sq + s) * HDim + hd)
              : ((((size_t)b * Hh + h) * HDim + hd) * Sq + s);
          outb[idx] = f2bf(v);
        }
      }
}

// ------------------------------------------------------------- flash attention
// grid: (S/128, B*H); 8 waves, each wave owns 16 query rows.
// K/V tiles staged cooperatively via GLOBAL_LOAD_ASYNC_TO_LDS_B128.
__global__ void k_attn(const bf16_t* __restrict__ Qb, const bf16_t* __restrict__ Kb,
                       const bf16_t* __restrict__ VTb, bf16_t* __restrict__ Ob) {
  __shared__ alignas(16) bf16_t kbuf[64 * 64];      // 8 KB: [key_local][hd]
  __shared__ alignas(16) bf16_t vbuf[64 * 64];      // 8 KB: [hd][key_local]
  __shared__ alignas(16) bf16_t pbuf[8][16 * 64];   // per-wave P tile

  const int tid  = threadIdx.x;
  const int lane = tid & 31;
  const int w    = tid >> 5;
  const int rlo  = lane & 15;
  const int hi   = lane >> 4;
  const int khalfA = hi * 8;
  const int k16B   = hi * 16;

  const int bh        = blockIdx.y;             // b*H + h
  const int blockBase = blockIdx.x * 128;
  const int qBase     = blockBase + w * 16;
  const int kbEnd     = blockBase + 128;        // uniform causal bound for block

  const bf16_t* Qh = Qb  + (size_t)bh * Sq * HDim;
  const bf16_t* Kh = Kb  + (size_t)bh * Sq * HDim;
  const bf16_t* Vh = VTb + (size_t)bh * HDim * Sq;
  bf16_t* pb = pbuf[w];

  const v8f vzero = {0.f,0.f,0.f,0.f,0.f,0.f,0.f,0.f};
  v8f acc[4];
  for (int jo = 0; jo < 4; ++jo) acc[jo] = vzero;
  float mrow[8], lrow[8];
  for (int r = 0; r < 8; ++r) { mrow[r] = -3.0e38f; lrow[r] = 0.f; }

  // Q fragments for the two HD k-steps (HD = 64 = 2 x 32), reused all key blocks
  v16bf qf[2];
  qf[0] = load_frag_a(Qh + (size_t)(qBase + rlo) * HDim + 0,  khalfA);
  qf[1] = load_frag_a(Qh + (size_t)(qBase + rlo) * HDim + 32, khalfA);

  const float scale = 0.125f;  // 1/sqrt(64)

  for (int kb = 0; kb < kbEnd; kb += 64) {
    // ---- cooperative async staging of K (8 KB contiguous) and V^T tiles
    __syncthreads();                            // previous iter's reads done
    {
      const bf16_t* gk = Kh + (size_t)kb * HDim;   // 64 rows x 128 B, contiguous
      for (int c = 0; c < 2; ++c) {
        int ch = tid + c * 256;                    // chunk id 0..511 (16 B each)
        async_ld16((unsigned)(size_t)(kbuf + ch * 8), gk + ch * 8);
        int hd = ch >> 3, co = (ch & 7) * 8;       // V^T: 64 rows stride Sq
        async_ld16((unsigned)(size_t)(vbuf + hd * 64 + co),
                   Vh + (size_t)hd * Sq + kb + co);
      }
    }
    wait_async0();
    __syncthreads();                            // tiles visible to all waves

    if (kb < qBase + 16) {                      // this wave's causal region
      // ---- S = Q K^T (16 x 64 tile, f32 accum), K from LDS
      v8f s[4];
      for (int j = 0; j < 4; ++j) s[j] = vzero;
      for (int kk = 0; kk < 2; ++kk)
        for (int j = 0; j < 4; ++j) {
          v16bf kf = load_frag_b(kbuf + (j * 16 + rlo) * 64 + kk * 32, k16B);
          s[j] = wmma_bf16(qf[kk], kf, s[j]);
        }

      // ---- scale + causal mask + online softmax (half-wave row reductions)
      for (int r = 0; r < 8; ++r) {
        int qrow = qBase + r + hi * 8;
        float rmax = -3.0e38f;
        for (int j = 0; j < 4; ++j) {
          int key = kb + j * 16 + rlo;
          float v = s[j][r] * scale;
          v = (key <= qrow) ? v : -3.0e38f;
          s[j][r] = v;
          rmax = fmaxf(rmax, v);
        }
        for (int off = 1; off < 16; off <<= 1)
          rmax = fmaxf(rmax, __shfl_xor(rmax, off, 32));
        float mnew  = fmaxf(mrow[r], rmax);
        float alpha = __expf(mrow[r] - mnew);
        float rsum  = 0.f;
        for (int j = 0; j < 4; ++j) {
          float p = __expf(s[j][r] - mnew);
          s[j][r] = p;
          rsum += p;
        }
        for (int off = 1; off < 16; off <<= 1)
          rsum += __shfl_xor(rsum, off, 32);
        lrow[r] = lrow[r] * alpha + rsum;
        mrow[r] = mnew;
        for (int jo = 0; jo < 4; ++jo) acc[jo][r] = acc[jo][r] * alpha;
      }

      // ---- transpose P from C-layout to A-layout via wave-private LDS tile
      for (int j = 0; j < 4; ++j)
        for (int r = 0; r < 8; ++r)
          pb[(r + hi * 8) * 64 + j * 16 + rlo] = f2bf(s[j][r]);

      // ---- O += P V  (contraction over 64 keys = 2 x 32), V from LDS
      for (int kk = 0; kk < 2; ++kk) {
        v16bf pf = load_frag_a(pb + rlo * 64 + kk * 32, khalfA);
        for (int jo = 0; jo < 4; ++jo) {
          v16bf vf = load_frag_b(vbuf + (jo * 16 + rlo) * 64 + kk * 32, k16B);
          acc[jo] = wmma_bf16(pf, vf, acc[jo]);
        }
      }
    }
  }

  // ---- epilogue: O / l, write bf16 [B,S,H*HD]
  const int b = bh >> 4;      // / Hh
  const int h = bh & 15;
  for (int jo = 0; jo < 4; ++jo)
    for (int r = 0; r < 8; ++r) {
      int q = qBase + r + hi * 8;
      float o = acc[jo][r] / lrow[r];
      Ob[((size_t)(b * Sq + q)) * Dm + h * HDim + jo * 16 + rlo] = f2bf(o);
    }
}

// ---------------------------------------------------------------- launcher
extern "C" void kernel_launch(void* const* d_in, const int* in_sizes, int n_in,
                              void* d_out, int out_size, void* d_ws, size_t ws_size,
                              hipStream_t stream) {
  (void)in_sizes; (void)n_in; (void)out_size; (void)ws_size;

  const float* x  = (const float*)d_in[0];
  const float* Wq = (const float*)d_in[1];
  const float* Wk = (const float*)d_in[2];
  const float* Wv = (const float*)d_in[3];
  const float* Wo = (const float*)d_in[4];
  const float* bo = (const float*)d_in[5];
  float* out = (float*)d_out;

  char* ws = (char*)d_ws;
  const size_t MB = (size_t)1 << 20;
  bf16_t* xb  = (bf16_t*)(ws + 0 * MB);    // 8 MB  [4096,1024]
  bf16_t* wqT = (bf16_t*)(ws + 8 * MB);    // 2 MB  [N,K]
  bf16_t* wkT = (bf16_t*)(ws + 10 * MB);
  bf16_t* wvT = (bf16_t*)(ws + 12 * MB);
  bf16_t* woT = (bf16_t*)(ws + 14 * MB);
  bf16_t* Qb  = (bf16_t*)(ws + 16 * MB);   // 8 MB  [B,H,S,HD]
  bf16_t* Kb  = (bf16_t*)(ws + 24 * MB);   // 8 MB  [B,H,S,HD]
  bf16_t* VTb = (bf16_t*)(ws + 32 * MB);   // 8 MB  [B,H,HD,S]
  bf16_t* Ob  = (bf16_t*)(ws + 40 * MB);   // 8 MB  [4096,1024]

  // 1) fp32 -> bf16 conversions / weight transposes
  int n = Mrows * Dm;
  k_cvt<<<n / 1024, 256, 0, stream>>>(x, xb, n);
  dim3 tb(32, 8), tg(Dm / 32, Dm / 32);
  k_transpose_cvt<<<tg, tb, 0, stream>>>(Wq, wqT);
  k_transpose_cvt<<<tg, tb, 0, stream>>>(Wk, wkT);
  k_transpose_cvt<<<tg, tb, 0, stream>>>(Wv, wvT);
  k_transpose_cvt<<<tg, tb, 0, stream>>>(Wo, woT);

  // 2) QKV projections (V stored pre-transposed for the P*V B-fragments)
  dim3 gg(Dm / 64, Mrows / 128);           // (16, 32), 256 threads
  k_gemm_bf16<<<gg, 256, 0, stream>>>(xb, wqT, Qb,  nullptr, nullptr, 0);
  k_gemm_bf16<<<gg, 256, 0, stream>>>(xb, wkT, Kb,  nullptr, nullptr, 0);
  k_gemm_bf16<<<gg, 256, 0, stream>>>(xb, wvT, VTb, nullptr, nullptr, 1);

  // 3) causal flash attention
  dim3 ag(Sq / 128, Bq * Hh);              // (16, 32)
  k_attn<<<ag, 256, 0, stream>>>(Qb, Kb, VTb, Ob);

  // 4) output projection + bias (fp32 out)
  k_gemm_bf16<<<gg, 256, 0, stream>>>(Ob, woT, nullptr, out, bo, 2);
}
